// DeformableBlock_62096637165753
// MI455X (gfx1250) — matrix operand order repaired and checked by reading, not compiled, not run on hardware
//
#include <hip/hip_runtime.h>
#include <hip/hip_bf16.h>

// Problem constants (from reference)
#define BB   8
#define CC   64
#define HH   128
#define WW   128
#define HW   (HH * WW)          // 16384
#define OFFC 18                 // 2*KK offset channels

typedef __attribute__((ext_vector_type(16))) __bf16 v16bf;
typedef __attribute__((ext_vector_type(8)))  float  v8f;

// ---- workspace layout (all offsets multiples of 256 B) ----
#define XT_OFF       ((size_t)0)
#define XT_BYTES     ((size_t)BB * HH * WW * CC * 2)      // 16,777,216 (bf16 NHWC)
#define OFFBUF_OFF   (XT_OFF + XT_BYTES)
#define OFFBUF_BYTES ((size_t)BB * OFFC * HW * 4)         //  9,437,184
#define APACK_OFF    (OFFBUF_OFF + OFFBUF_BYTES)
#define APACK_BYTES  ((size_t)4 * 18 * 32 * 16 * 2)       //     73,728
#define AOFF_OFF     (APACK_OFF + APACK_BYTES)
#define AOFF_BYTES   ((size_t)2 * 50 * 32 * 16 * 2)       //    102,400

__device__ __forceinline__ unsigned short f2bf(float f) {
    union { float f; unsigned u; } cv; cv.f = f;
    unsigned u = cv.u;
    unsigned r = u + 0x7FFFu + ((u >> 16) & 1u);   // round-to-nearest-even
    return (unsigned short)(r >> 16);
}
__device__ __forceinline__ unsigned bfsplat(float f) {
    unsigned b = (unsigned)f2bf(f);
    return b | (b << 16);
}
__device__ __forceinline__ int iclamp(int v, int lo, int hi) {
    return v < lo ? lo : (v > hi ? hi : v);
}
// Packed bf16 math: guarantee the VOP3P path (headers may promote via f32)
__device__ __forceinline__ unsigned pk_mul_bf16(unsigned a, unsigned b) {
    unsigned d;
    asm volatile("v_pk_mul_bf16 %0, %1, %2" : "=v"(d) : "v"(a), "v"(b));
    return d;
}
__device__ __forceinline__ unsigned pk_fma_bf16(unsigned a, unsigned b, unsigned c) {
    unsigned d;
    asm volatile("v_pk_fma_bf16 %0, %1, %2, %3" : "=v"(d) : "v"(a), "v"(b), "v"(c));
    return d;
}

// ------------------------------------------------------------------
// Kernel 0: NCHW f32 -> NHWC bf16 transpose via LDS tile (64c x 64x)
// ------------------------------------------------------------------
__global__ void xpose_kernel(const float* __restrict__ x, unsigned short* __restrict__ xTb) {
    __shared__ float tile[64][65];
    int bid  = blockIdx.x;
    int x0   = (bid & 1) * 64;
    int row  = bid >> 1;          // row = b*128 + y
    int tid  = threadIdx.x;
#pragma unroll
    for (int rep = 0; rep < 16; ++rep) {
        int idx = rep * 256 + tid;
        int c = idx >> 6, xx = idx & 63;
        // x[b][c][y][x0+xx] ; (b*64+c)*HH*WW + y*WW + .. == (b<<6|c)<<14 | y<<7 | ..
        unsigned b = (unsigned)row >> 7, y = (unsigned)row & 127;
        tile[c][xx] = x[(((b << 6) + c) << 14) + (y << 7) + x0 + xx];
    }
    __syncthreads();
#pragma unroll
    for (int rep = 0; rep < 16; ++rep) {
        int idx = rep * 256 + tid;
        int xx = idx >> 6, c = idx & 63;
        xTb[(((unsigned)row << 7) + x0 + xx) * CC + c] = f2bf(tile[c][xx]);
    }
}

// ------------------------------------------------------------------
// Kernel 1: pack main conv weights (64,64,3,3) into WMMA A-fragment
// layout: aPack[(mtile*18 + kchunk)*32 + lane][16] bf16
// K order: K = t*64 + c ; kchunk = t*2 + kb covers c = kb*32 + klocal
// A layout used here: lane L row M=L&15, klocal = (L>>4)*16 + {0..15}
// ------------------------------------------------------------------
__global__ void pack_main_w(const float* __restrict__ w, unsigned short* __restrict__ aP) {
    int id = blockIdx.x * 256 + threadIdx.x;
    if (id >= 4 * 18 * 32 * 16) return;
    int h = id & 15;   int rest = id >> 4;
    int lane = rest & 31; rest >>= 5;
    int kchunk = rest % 18; int mtile = rest / 18;
    int klocal = ((lane >> 4) << 4) + h;
    int c = ((kchunk & 1) << 5) + klocal;
    int t = kchunk >> 1;                      // tap 0..8
    int M = (mtile << 4) + (lane & 15);       // output channel 0..63
    float v = w[(unsigned)((M << 6) + c) * 9 + t];
    aP[id] = f2bf(v);
}

// ------------------------------------------------------------------
// Kernel 2: pack offset conv weights (18,64,5,5), M padded to 32
// kchunk = tap*2 + kb, tap 0..24 ; aOff[(mtile*50+kchunk)*32+lane][16]
// ------------------------------------------------------------------
__global__ void pack_off_w(const float* __restrict__ ow, unsigned short* __restrict__ aP) {
    int id = blockIdx.x * 256 + threadIdx.x;
    if (id >= 2 * 50 * 32 * 16) return;
    int h = id & 15;   int rest = id >> 4;
    int lane = rest & 31; rest >>= 5;
    int kchunk = rest % 50; int mtile = rest / 50;
    int klocal = ((lane >> 4) << 4) + h;
    int c = ((kchunk & 1) << 5) + klocal;
    int tap = kchunk >> 1;
    int i = tap / 5, j = tap % 5;
    int M = (mtile << 4) + (lane & 15);
    float v = (M < OFFC) ? ow[(unsigned)(((M << 6) + c) * 5 + i) * 5 + j] : 0.0f;
    aP[id] = f2bf(v);
}

// ------------------------------------------------------------------
// Kernel 3: offset conv as implicit GEMM (M=32 padded, K=1600, N=16/wave)
// B-fragment is a masked copy of bf16 NHWC data (no conversion).
// bias + clip fused; output layout (b, 18, h, w) f32
// All addressing via 32-bit element offsets (SADDR + voffset form).
// ------------------------------------------------------------------
__global__ void offconv_kernel(const unsigned short* __restrict__ xTb,
                               const unsigned short* __restrict__ aOffPack,
                               const float* __restrict__ bias,
                               float* __restrict__ offBuf) {
    int lane = threadIdx.x & 31;
    int wav  = threadIdx.x >> 5;
    int tile = blockIdx.x * 8 + wav;       // 8192 tiles total
    int n    = lane & 15;
    unsigned g   = (unsigned)tile * 16 + n;   // global pixel
    unsigned b   = g >> 14;
    unsigned rem = g & 16383;
    int y  = (int)(rem >> 7);
    int xq = (int)(rem & 127);
    int half = lane >> 4;

    const v8f vzero = {0.f,0.f,0.f,0.f,0.f,0.f,0.f,0.f};
    v8f acc0 = vzero, acc1 = vzero;
    const v16bf* aP = (const v16bf*)aOffPack;
    unsigned brow = b << 7;                   // b*HH

    for (int tap = 0; tap < 25; ++tap) {
        int i = tap / 5, j = tap % 5;
        int ys = y + i - 2, xs = xq + j - 2;
        unsigned msk = (ys >= 0 && ys < HH && xs >= 0 && xs < WW) ? 0xFFFFFFFFu : 0u;
        unsigned ysc = (unsigned)iclamp(ys, 0, HH - 1);
        unsigned xsc = (unsigned)iclamp(xs, 0, WW - 1);
        unsigned pofs = ((((brow + ysc) << 7) + xsc) << 6);   // element offset
#pragma unroll
        for (int kb = 0; kb < 2; ++kb) {
            unsigned c0 = ((unsigned)kb << 5) + ((unsigned)half << 4);
            const uint4* q = (const uint4*)(xTb + pofs + c0);
            uint4 ra = q[0], rb = q[1];
            union { v16bf v; unsigned u[8]; } bu;
            bu.u[0] = ra.x & msk; bu.u[1] = ra.y & msk;
            bu.u[2] = ra.z & msk; bu.u[3] = ra.w & msk;
            bu.u[4] = rb.x & msk; bu.u[5] = rb.y & msk;
            bu.u[6] = rb.z & msk; bu.u[7] = rb.w & msk;
            unsigned kchunk = (unsigned)tap * 2 + kb;
            v16bf a0 = aP[(kchunk << 5) + lane];
            v16bf a1 = aP[((50u + kchunk) << 5) + lane];
            acc0 = __builtin_amdgcn_wmma_f32_16x16x32_bf16(false, a0, false, bu.v,
                                                           (short)0, acc0, false, false);
            acc1 = __builtin_amdgcn_wmma_f32_16x16x32_bf16(false, a1, false, bu.v,
                                                           (short)0, acc1, false, false);
        }
    }
    // C/D layout: VGPR r, N = lane&15, M = r + 8*half (+16 for acc1)
    unsigned obase = b * (OFFC * HW) + rem;
#pragma unroll
    for (int r = 0; r < 8; ++r) {
        unsigned ch = (unsigned)r + ((unsigned)half << 3);
        float v = acc0[r] + bias[ch];
        v = fminf(fmaxf(v, -1.0f), 1.0f);
        offBuf[obase + ch * HW] = v;
    }
    if (half == 0) {
#pragma unroll
        for (int r = 0; r < 2; ++r) {
            unsigned ch = 16u + r;             // only 16,17 are real channels
            float v = acc1[r] + bias[ch];
            v = fminf(fmaxf(v, -1.0f), 1.0f);
            offBuf[obase + ch * HW] = v;
        }
    }
}

// ------------------------------------------------------------------
// Kernel 4: fused bilinear sampling (packed bf16) + deform GEMM
// M=64 (4 tiles), K=576 (18 chunks of 32: K = t*64 + c), N=16 px/wave
// ------------------------------------------------------------------
__global__ void deform_kernel(const unsigned short* __restrict__ xTb,
                              const float* __restrict__ offBuf,
                              const unsigned short* __restrict__ aPack,
                              float* __restrict__ out) {
    int lane = threadIdx.x & 31;
    int wav  = threadIdx.x >> 5;
    int tile = blockIdx.x * 8 + wav;
    int n    = lane & 15;
    unsigned g   = (unsigned)tile * 16 + n;
    unsigned b   = g >> 14;
    unsigned rem = g & 16383;
    int y  = (int)(rem >> 7);
    int xq = (int)(rem & 127);
    int half = lane >> 4;

    const v8f vzero = {0.f,0.f,0.f,0.f,0.f,0.f,0.f,0.f};
    v8f acc[4];
#pragma unroll
    for (int m = 0; m < 4; ++m) acc[m] = vzero;
    const v16bf* aP = (const v16bf*)aPack;
    unsigned brow  = b << 7;                      // b*HH
    unsigned offb  = b * (18u * HW) + rem;        // offBuf base for this pixel

    for (int t = 0; t < 9; ++t) {
        int ti = t / 3, tj = t % 3;
        float dy = offBuf[offb + (unsigned)(t * 2 + 0) * HW];
        float dx = offBuf[offb + (unsigned)(t * 2 + 1) * HW];
        float py = (float)(y + ti - 1) + dy;
        float px = (float)(xq + tj - 1) + dx;
        float y0f = floorf(py), x0f = floorf(px);
        float fy = py - y0f, fx = px - x0f;
        int y0 = (int)y0f, x0 = (int)x0f;
        int y1 = y0 + 1,   x1 = x0 + 1;
        float vy0 = (y0 >= 0 && y0 < HH) ? 1.f : 0.f;
        float vy1 = (y1 >= 0 && y1 < HH) ? 1.f : 0.f;
        float vx0 = (x0 >= 0 && x0 < WW) ? 1.f : 0.f;
        float vx1 = (x1 >= 0 && x1 < WW) ? 1.f : 0.f;
        // bilinear weights (validity folded in), splatted to packed bf16
        unsigned w00v = bfsplat((1.f - fy) * (1.f - fx) * vy0 * vx0);
        unsigned w01v = bfsplat((1.f - fy) * fx         * vy0 * vx1);
        unsigned w10v = bfsplat(fy         * (1.f - fx) * vy1 * vx0);
        unsigned w11v = bfsplat(fy         * fx         * vy1 * vx1);
        unsigned y0c = (unsigned)iclamp(y0, 0, HH - 1), y1c = (unsigned)iclamp(y1, 0, HH - 1);
        unsigned x0c = (unsigned)iclamp(x0, 0, WW - 1), x1c = (unsigned)iclamp(x1, 0, WW - 1);
        unsigned row0 = (brow + y0c) << 7, row1 = (brow + y1c) << 7;
        unsigned o00 = (row0 + x0c) << 6, o01 = (row0 + x1c) << 6;
        unsigned o10 = (row1 + x0c) << 6, o11 = (row1 + x1c) << 6;
#pragma unroll
        for (int kb = 0; kb < 2; ++kb) {
            unsigned c0 = ((unsigned)kb << 5) + ((unsigned)half << 4);
            const uint4* q00 = (const uint4*)(xTb + o00 + c0);
            const uint4* q01 = (const uint4*)(xTb + o01 + c0);
            const uint4* q10 = (const uint4*)(xTb + o10 + c0);
            const uint4* q11 = (const uint4*)(xTb + o11 + c0);
            uint4 r00a = q00[0], r00b = q00[1];
            uint4 r01a = q01[0], r01b = q01[1];
            uint4 r10a = q10[0], r10b = q10[1];
            uint4 r11a = q11[0], r11b = q11[1];
            union { v16bf v; unsigned u[8]; } bu;
#define PKCOMB(dst, s00, s01, s10, s11)                                  \
            do {                                                         \
                unsigned t_ = pk_mul_bf16((s00), w00v);                  \
                t_ = pk_fma_bf16((s01), w01v, t_);                       \
                t_ = pk_fma_bf16((s10), w10v, t_);                       \
                (dst) = pk_fma_bf16((s11), w11v, t_);                    \
            } while (0)
            PKCOMB(bu.u[0], r00a.x, r01a.x, r10a.x, r11a.x);
            PKCOMB(bu.u[1], r00a.y, r01a.y, r10a.y, r11a.y);
            PKCOMB(bu.u[2], r00a.z, r01a.z, r10a.z, r11a.z);
            PKCOMB(bu.u[3], r00a.w, r01a.w, r10a.w, r11a.w);
            PKCOMB(bu.u[4], r00b.x, r01b.x, r10b.x, r11b.x);
            PKCOMB(bu.u[5], r00b.y, r01b.y, r10b.y, r11b.y);
            PKCOMB(bu.u[6], r00b.z, r01b.z, r10b.z, r11b.z);
            PKCOMB(bu.u[7], r00b.w, r01b.w, r10b.w, r11b.w);
#undef PKCOMB
            unsigned kchunk = (unsigned)t * 2 + kb;
#pragma unroll
            for (int m = 0; m < 4; ++m) {
                v16bf a = aP[(((unsigned)m * 18 + kchunk) << 5) + lane];
                acc[m] = __builtin_amdgcn_wmma_f32_16x16x32_bf16(false, a, false, bu.v,
                                                                 (short)0, acc[m], false, false);
            }
        }
    }
    unsigned obase = (b << 20) + rem;            // b*CC*HW + rem
#pragma unroll
    for (int m = 0; m < 4; ++m) {
#pragma unroll
        for (int r = 0; r < 8; ++r) {
            unsigned o = ((unsigned)m << 4) + r + ((unsigned)half << 3);
            out[obase + (o << 14)] = acc[m][r];
        }
    }
}

extern "C" void kernel_launch(void* const* d_in, const int* in_sizes, int n_in,
                              void* d_out, int out_size, void* d_ws, size_t ws_size,
                              hipStream_t stream) {
    const float* x        = (const float*)d_in[0];   // (8,64,128,128)
    const float* weights  = (const float*)d_in[1];   // (64,64,3,3)
    const float* offset_w = (const float*)d_in[2];   // (18,64,5,5)
    const float* offset_b = (const float*)d_in[3];   // (18,)
    float* out = (float*)d_out;

    char* ws = (char*)d_ws;
    unsigned short* xTb    = (unsigned short*)(ws + XT_OFF);
    float*          offBuf = (float*)(ws + OFFBUF_OFF);
    unsigned short* aPack  = (unsigned short*)(ws + APACK_OFF);
    unsigned short* aOff   = (unsigned short*)(ws + AOFF_OFF);

    xpose_kernel<<<2048, 256, 0, stream>>>(x, xTb);
    pack_main_w<<<(4 * 18 * 32 * 16 + 255) / 256, 256, 0, stream>>>(weights, aPack);
    pack_off_w <<<(2 * 50 * 32 * 16 + 255) / 256, 256, 0, stream>>>(offset_w, aOff);
    offconv_kernel<<<1024, 256, 0, stream>>>(xTb, aOff, offset_b, offBuf);
    deform_kernel <<<1024, 256, 0, stream>>>(xTb, offBuf, aPack, out);
}
